// RelMultiHeadAttn_6837587936064
// MI455X (gfx1250) — matrix operand order repaired and checked by reading, not compiled, not run on hardware
//
#include <hip/hip_runtime.h>
#include <hip/hip_bf16.h>

typedef __attribute__((ext_vector_type(8)))  _Float16 v8h;
typedef __attribute__((ext_vector_type(16))) _Float16 v16h;
typedef __attribute__((ext_vector_type(8)))  float    v8f;

#define QLEN   1024
#define BSZ    2
#define DMODEL 1024
#define NH     16
#define DHD    64
#define ATTN_SCALE 0.125f   // 1/sqrt(64)

// ---------------------------------------------------------------------------
// WMMA fragment loaders (wave32, v_wmma_f32_16x16x32_f16)
//
// A (16x32 f16): lane L: M = L&15; halves 0..7  -> K = (L>>4)*8 + h
//                                 halves 8..15 -> K = 16 + (L>>4)*8 + (h-8)
// B (32x16 f16): lane L: N = L&15; halves 0..7  -> K = (L>>4)*8 + h
//                                 halves 8..15 -> K = 16 + (L>>4)*8 + (h-8)
//   (B stored transposed as [N][K] row-major so each lane's 16 K-values are two contiguous runs of 8)
// C/D (16x16 f32): VGPR r: lanes 0-15 -> M=r, lanes 16-31 -> M=r+8; N = lane&15
// ---------------------------------------------------------------------------
static __device__ __forceinline__ v16h load_a16x32(const _Float16* tile, int ld, int lane) {
  const int m  = lane & 15;
  const int hi = lane >> 4;
  const _Float16* p = tile + (size_t)m * ld + hi * 8;
  v8h lo = *(const v8h*)p;
  v8h hh = *(const v8h*)(p + 16);
  v16h r;
#pragma unroll
  for (int i = 0; i < 8; ++i) { r[i] = lo[i]; r[i + 8] = hh[i]; }
  return r;
}

static __device__ __forceinline__ v16h load_bT32x16(const _Float16* tileT, int ldT, int lane) {
  const int n  = lane & 15;
  const int kh = lane >> 4;
  const _Float16* p = tileT + (size_t)n * ldT + kh * 8;
  v8h lo = *(const v8h*)p;
  v8h hh = *(const v8h*)(p + 16);
  v16h r;
#pragma unroll
  for (int i = 0; i < 8; ++i) { r[i] = lo[i]; r[i + 8] = hh[i]; }
  return r;
}

#define WMMA_F16(a, b, c) \
  __builtin_amdgcn_wmma_f32_16x16x32_f16(false, (a), false, (b), (short)0, (c), false, false)

// ---------------------------------------------------------------------------
// Elementwise f32 -> f16
// ---------------------------------------------------------------------------
__global__ void cvt_f16_kernel(const float* __restrict__ in, _Float16* __restrict__ out, int n) {
  int i = blockIdx.x * 256 + threadIdx.x;
  if (i < n) out[i] = (_Float16)in[i];
}

// in [K][N] f32 -> out [N][K] f16 (transpose so GEMM B-fragments are contiguous)
__global__ void cvt_f16_T_kernel(const float* __restrict__ in, _Float16* __restrict__ out,
                                 int K, int N) {
  int i = blockIdx.x * 256 + threadIdx.x;
  if (i >= K * N) return;
  int k = i / N, n = i - k * N;
  out[(size_t)n * K + k] = (_Float16)in[i];
}

// ---------------------------------------------------------------------------
// C[M,N] f32 = A[M,K] f16 (row major) x Bt[N,K] f16 (B transposed)
// block = 256 threads = 8 waves; each wave computes a 16x64 C tile
// (one A fragment feeds 4 WMMAs); K-loop register double-buffered so the
// 4 MMAs of chunk k overlap the 5 fragment loads of chunk k+32.
// ---------------------------------------------------------------------------
__global__ void gemm_f16_wmma(const _Float16* __restrict__ A, const _Float16* __restrict__ Bt,
                              float* __restrict__ C, int M, int N, int K) {
  const int lane = threadIdx.x & 31;
  const int wave = threadIdx.x >> 5;
  const int tn4 = blockIdx.x * 8 + wave;        // 64-column group index
  const int tm  = blockIdx.y;
  if (tn4 * 64 >= N || tm * 16 >= M) return;
  const _Float16* Arow = A + (size_t)tm * 16 * K;
  const _Float16* B0 = Bt + (size_t)(tn4 * 64 +  0) * K;
  const _Float16* B1 = Bt + (size_t)(tn4 * 64 + 16) * K;
  const _Float16* B2 = Bt + (size_t)(tn4 * 64 + 32) * K;
  const _Float16* B3 = Bt + (size_t)(tn4 * 64 + 48) * K;

  v8f acc0 = {}, acc1 = {}, acc2 = {}, acc3 = {};
  v16h a  = load_a16x32(Arow, K, lane);
  v16h b0 = load_bT32x16(B0, K, lane);
  v16h b1 = load_bT32x16(B1, K, lane);
  v16h b2 = load_bT32x16(B2, K, lane);
  v16h b3 = load_bT32x16(B3, K, lane);

  for (int k0 = 32; k0 < K; k0 += 32) {
    if (k0 + 32 < K) {
      __builtin_prefetch(Arow + (size_t)(lane & 15) * K + k0 + 32, 0, 3);
      __builtin_prefetch(B0 + (size_t)(lane & 15) * K + k0 + 32, 0, 3);
    }
    v16h an = load_a16x32(Arow + k0, K, lane);
    v16h c0 = load_bT32x16(B0 + k0, K, lane);
    v16h c1 = load_bT32x16(B1 + k0, K, lane);
    v16h c2 = load_bT32x16(B2 + k0, K, lane);
    v16h c3 = load_bT32x16(B3 + k0, K, lane);
    acc0 = WMMA_F16(a, b0, acc0);
    acc1 = WMMA_F16(a, b1, acc1);
    acc2 = WMMA_F16(a, b2, acc2);
    acc3 = WMMA_F16(a, b3, acc3);
    a = an; b0 = c0; b1 = c1; b2 = c2; b3 = c3;
  }
  acc0 = WMMA_F16(a, b0, acc0);
  acc1 = WMMA_F16(a, b1, acc1);
  acc2 = WMMA_F16(a, b2, acc2);
  acc3 = WMMA_F16(a, b3, acc3);

  const int col = tn4 * 64 + (lane & 15);
  const int hi  = lane >> 4;
#pragma unroll
  for (int r = 0; r < 8; ++r) {
    const size_t row = (size_t)(tm * 16 + r + 8 * hi) * N;
    C[row + col +  0] = acc0[r];
    C[row + col + 16] = acc1[r];
    C[row + col + 32] = acc2[r];
    C[row + col + 48] = acc3[r];
  }
}

// ---------------------------------------------------------------------------
// Split heads from QKV output [Q*B, 3*N*DH] into per-head f16 tensors:
//   q1 = q + r_w_bias, q2 = q + r_r_bias, k  : layout [B*N][Q][DH]
//   vT : layout [B*N][DH][Q]  (transposed so PV B-fragments are contiguous)
// ---------------------------------------------------------------------------
__global__ void prep_qkv_kernel(const float* __restrict__ heads,
                                const float* __restrict__ rwb, const float* __restrict__ rrb,
                                _Float16* __restrict__ q1, _Float16* __restrict__ q2,
                                _Float16* __restrict__ kh, _Float16* __restrict__ vT) {
  int idx = blockIdx.x * 256 + threadIdx.x;          // [0, 2048*1024)
  int qb = idx >> 10;
  int c  = idx & 1023;                               // c = n*64 + d
  int q = qb >> 1, b = qb & 1;
  int n = c >> 6,  d = c & 63;
  size_t hrow = (size_t)qb * (3 * DMODEL);
  float qv = heads[hrow + c];
  float kv = heads[hrow + DMODEL + c];
  float vv = heads[hrow + 2 * DMODEL + c];
  size_t o = ((size_t)(b * NH + n) * QLEN + q) * DHD + d;
  q1[o] = (_Float16)(qv + rwb[c]);
  q2[o] = (_Float16)(qv + rrb[c]);
  kh[o] = (_Float16)kv;
  vT[((size_t)(b * NH + n) * DHD + d) * QLEN + q] = (_Float16)vv;
}

// r_k[k] = r_fw[k] (k<Q) | r_bw[2Q-2-k] (Q<=k<2Q-1) ; row 2047 = zero pad.
// Layout [N][2048][DH] f16.
__global__ void prep_rk_kernel(const float* __restrict__ rfw, const float* __restrict__ rbw,
                               _Float16* __restrict__ rk) {
  int idx = blockIdx.x * 256 + threadIdx.x;          // [0, 16*2048*64)
  int d = idx & 63;
  int rest = idx >> 6;
  int k = rest & 2047;
  int n = rest >> 11;
  int c = n * DHD + d;
  float v = 0.f;
  if (k < QLEN)              v = rfw[(size_t)k * DMODEL + c];
  else if (k < 2 * QLEN - 1) v = rbw[(size_t)(2 * QLEN - 2 - k) * DMODEL + c];
  rk[idx] = (_Float16)v;
}

// ---------------------------------------------------------------------------
// Flash attention with Transformer-XL relative shift.
//   score(i,j) = (q1_i . k_j + q2_i . r_k[Q-1-i+j]) * SCALE  (then attn_mask)
// One wave per (b, n, 16-query tile); 32-key chunks; online softmax.
// BD term: 16x48 band GEMM + Toeplitz diagonal extraction through LDS.
// K tiles are async-copied global->LDS (per-wave ASYNCcnt double buffer):
// chunk j+32 streams in via global_load_async_to_lds_b128 while the AC/BD/PV
// WMMAs of chunk j execute out of LDS.
// ---------------------------------------------------------------------------
__global__ void attn_flash_kernel(const _Float16* __restrict__ q1,
                                  const _Float16* __restrict__ q2,
                                  const _Float16* __restrict__ kh,
                                  const _Float16* __restrict__ vT,
                                  const _Float16* __restrict__ rk,
                                  const unsigned char* __restrict__ mask,
                                  _Float16* __restrict__ vec) {
  __shared__ float bandP[8][16][48];
  __shared__ __align__(16) _Float16 pf16[8][16][32];
  __shared__ __align__(16) _Float16 karr[8][2][32 * DHD];   // 2 x 4KB per wave

  const int lane = threadIdx.x & 31;
  const int wave = threadIdx.x >> 5;
  const int task = blockIdx.x * 8 + wave;            // [0, B*NH*(Q/16)) = 2048
  const int it = task & 63;
  const int bn = task >> 6;                          // b*NH + n
  const int b  = bn >> 4;
  const int n  = bn & 15;
  const int i0 = it * 16;
  const int nn = lane & 15;
  const int hi = lane >> 4;

  const _Float16* qb1 = q1 + ((size_t)bn * QLEN + i0) * DHD;
  const _Float16* qb2 = q2 + ((size_t)bn * QLEN + i0) * DHD;
  const v16h a1_0 = load_a16x32(qb1,      DHD, lane);
  const v16h a1_1 = load_a16x32(qb1 + 32, DHD, lane);
  const v16h a2_0 = load_a16x32(qb2,      DHD, lane);
  const v16h a2_1 = load_a16x32(qb2 + 32, DHD, lane);

  const _Float16* kbase = kh + (size_t)bn * QLEN * DHD;
  const _Float16* vbase = vT + (size_t)bn * DHD * QLEN;
  const _Float16* rbase = rk + (size_t)n * 2048 * DHD;

  // per-wave async K-tile stage: 32 rows x 64 halves = 4KB contiguous
  const unsigned klds0 = (unsigned)(size_t)(void*)&karr[wave][0][0] + (unsigned)lane * 16u;
  const unsigned klds1 = (unsigned)(size_t)(void*)&karr[wave][1][0] + (unsigned)lane * 16u;
  auto issueK = [&](int j0, int buf) {
    const _Float16* g = kbase + (size_t)j0 * DHD + lane * 8;   // 16B per lane
    unsigned l = buf ? klds1 : klds0;
#pragma unroll
    for (int i = 0; i < 8; ++i) {
      asm volatile("global_load_async_to_lds_b128 %0, %1, off"
                   :: "v"(l + (unsigned)(i * 512)), "v"(g + i * 256)
                   : "memory");
    }
  };
  issueK(0, 0);

  float mrow[8], lrow[8];
#pragma unroll
  for (int r = 0; r < 8; ++r) { mrow[r] = -1e30f; lrow[r] = 0.f; }
  v8f acc0 = {}, acc1 = {}, acc2 = {}, acc3 = {};

  for (int j0 = 0; j0 < QLEN; j0 += 32) {
    const int buf = (j0 >> 5) & 1;
    // current K tile landed in LDS; next tile streams while we compute
    asm volatile("s_wait_asynccnt 0x0" ::: "memory");
    if (j0 + 32 < QLEN) issueK(j0 + 32, buf ^ 1);

    // ---- AC: (q + r_w_bias) . K^T, two 16x16 tiles, K from LDS ----
    v8f S0 = {}, S1 = {};
    {
      const _Float16* kt0 = &karr[wave][buf][0];
      S0 = WMMA_F16(a1_0, load_bT32x16(kt0,      DHD, lane), S0);
      S0 = WMMA_F16(a1_1, load_bT32x16(kt0 + 32, DHD, lane), S0);
      const _Float16* kt1 = kt0 + 16 * DHD;
      S1 = WMMA_F16(a1_0, load_bT32x16(kt1,      DHD, lane), S1);
      S1 = WMMA_F16(a1_1, load_bT32x16(kt1 + 32, DHD, lane), S1);
    }
    // ---- BD band: P[di, t] = q2 . r_k[base + t], t in [0,48) ----
    const int base = QLEN - 16 - i0 + j0;            // >= 0; base+47 <= 2047 (pad row)
#pragma unroll
    for (int bt = 0; bt < 3; ++bt) {
      v8f pb = {};
      const _Float16* rt = rbase + (size_t)(base + bt * 16) * DHD;
      pb = WMMA_F16(a2_0, load_bT32x16(rt,      DHD, lane), pb);
      pb = WMMA_F16(a2_1, load_bT32x16(rt + 32, DHD, lane), pb);
#pragma unroll
      for (int r = 0; r < 8; ++r)
        bandP[wave][r + 8 * hi][bt * 16 + nn] = pb[r];
    }
    // LDS ops from the same wave are in-order: safe to read the band back.
    float p0[8], p1[8], alpha[8];
#pragma unroll
    for (int r = 0; r < 8; ++r) {
      const int row = r + 8 * hi;                    // di
      // BD_shifted[di, dj] = band[di][15 - di + dj]
      float s0 = (S0[r] + bandP[wave][row][15 - row + nn]) * ATTN_SCALE;
      float s1 = (S1[r] + bandP[wave][row][31 - row + nn]) * ATTN_SCALE;
      if (mask[(size_t)(j0 + nn) * BSZ + b])      s0 = -1e30f;
      if (mask[(size_t)(j0 + 16 + nn) * BSZ + b]) s1 = -1e30f;
      float tmax = fmaxf(s0, s1);
#pragma unroll
      for (int m = 1; m <= 8; m <<= 1) tmax = fmaxf(tmax, __shfl_xor(tmax, m, 32));
      const float mnew = fmaxf(mrow[r], tmax);
      alpha[r] = __expf(mrow[r] - mnew);
      p0[r] = __expf(s0 - mnew);
      p1[r] = __expf(s1 - mnew);
      float rs = p0[r] + p1[r];
#pragma unroll
      for (int m = 1; m <= 8; m <<= 1) rs += __shfl_xor(rs, m, 32);
      lrow[r] = lrow[r] * alpha[r] + rs;
      mrow[r] = mnew;
      pf16[wave][row][nn]      = (_Float16)p0[r];
      pf16[wave][row][16 + nn] = (_Float16)p1[r];
    }
#pragma unroll
    for (int r = 0; r < 8; ++r) {
      acc0[r] *= alpha[r]; acc1[r] *= alpha[r];
      acc2[r] *= alpha[r]; acc3[r] *= alpha[r];
    }
    // ---- P (C layout) -> A fragment via LDS ----
    v16h pa;
    {
      const _Float16* pp = &pf16[wave][nn][hi * 8];
#pragma unroll
      for (int i = 0; i < 8; ++i) { pa[i] = pp[i]; pa[i + 8] = pp[16 + i]; }
    }
    // ---- PV: acc[t] += P[16x32] x V[32x16] per 16-wide d chunk ----
    acc0 = WMMA_F16(pa, load_bT32x16(vbase + (size_t)(0 * 16) * QLEN + j0, QLEN, lane), acc0);
    acc1 = WMMA_F16(pa, load_bT32x16(vbase + (size_t)(1 * 16) * QLEN + j0, QLEN, lane), acc1);
    acc2 = WMMA_F16(pa, load_bT32x16(vbase + (size_t)(2 * 16) * QLEN + j0, QLEN, lane), acc2);
    acc3 = WMMA_F16(pa, load_bT32x16(vbase + (size_t)(3 * 16) * QLEN + j0, QLEN, lane), acc3);
  }

  // ---- normalize and write vec_f16 [Q*B][N*DH] for the output projection ----
#pragma unroll
  for (int r = 0; r < 8; ++r) {
    const float inv = 1.f / lrow[r];
    const int row = r + 8 * hi;
    const size_t orow = ((size_t)(i0 + row) * BSZ + b) * (NH * DHD) + (size_t)n * DHD;
    vec[orow +  0 + nn] = (_Float16)(acc0[r] * inv);
    vec[orow + 16 + nn] = (_Float16)(acc1[r] * inv);
    vec[orow + 32 + nn] = (_Float16)(acc2[r] * inv);
    vec[orow + 48 + nn] = (_Float16)(acc3[r] * inv);
  }
}

// ---------------------------------------------------------------------------
// out = LayerNorm(w + attn_out) * g + b     (one block per (q,b) row)
// ---------------------------------------------------------------------------
__global__ void add_ln_kernel(const float* __restrict__ w, const float* __restrict__ attn,
                              const float* __restrict__ g, const float* __restrict__ bet,
                              float* __restrict__ out) {
  const int row = blockIdx.x;
  const int t = threadIdx.x;
  __shared__ float s1[256], s2[256];
  const float* wr = w    + (size_t)row * DMODEL;
  const float* ar = attn + (size_t)row * DMODEL;
  float x[4];
  float a = 0.f, b2 = 0.f;
#pragma unroll
  for (int j = 0; j < 4; ++j) {
    int i = t + j * 256;
    x[j] = wr[i] + ar[i];
    a += x[j]; b2 += x[j] * x[j];
  }
  s1[t] = a; s2[t] = b2;
  __syncthreads();
  for (int off = 128; off > 0; off >>= 1) {
    if (t < off) { s1[t] += s1[t + off]; s2[t] += s2[t + off]; }
    __syncthreads();
  }
  const float mu  = s1[0] * (1.0f / DMODEL);
  const float var = s2[0] * (1.0f / DMODEL) - mu * mu;
  const float inv = rsqrtf(var + 1e-5f);
#pragma unroll
  for (int j = 0; j < 4; ++j) {
    int i = t + j * 256;
    out[(size_t)row * DMODEL + i] = (x[j] - mu) * inv * g[i] + bet[i];
  }
}

// ---------------------------------------------------------------------------
extern "C" void kernel_launch(void* const* d_in, const int* in_sizes, int n_in,
                              void* d_out, int out_size, void* d_ws, size_t ws_size,
                              hipStream_t stream) {
  (void)in_sizes; (void)n_in; (void)out_size; (void)ws_size;
  const float* w    = (const float*)d_in[0];   // [Q,B,D]
  const float* r    = (const float*)d_in[1];   // [Q,D]
  const float* rwb  = (const float*)d_in[2];   // [N,DH]
  const float* rrb  = (const float*)d_in[3];   // [N,DH]
  const unsigned char* mask = (const unsigned char*)d_in[4];  // [Q,B] bool
  const float* Wqkv = (const float*)d_in[5];   // [D, 3*N*DH]
  const float* Wrfw = (const float*)d_in[6];   // [D, N*DH]
  const float* Wrbw = (const float*)d_in[7];   // [D, N*DH]
  const float* Wo   = (const float*)d_in[8];   // [N*DH, D]
  const float* lng  = (const float*)d_in[9];
  const float* lnb  = (const float*)d_in[10];
  float* out = (float*)d_out;

  char* ws = (char*)d_ws;
  size_t off = 0;
  auto alloc = [&](size_t bytes) -> void* {
    void* p = ws + off;
    off = (off + bytes + 255) & ~(size_t)255;
    return p;
  };

  _Float16* wf16  = (_Float16*)alloc(2048ull * 1024 * 2);
  _Float16* rf16  = (_Float16*)alloc(1024ull * 1024 * 2);
  _Float16* WqkvT = (_Float16*)alloc(3072ull * 1024 * 2);
  _Float16* WrfwT = (_Float16*)alloc(1024ull * 1024 * 2);
  _Float16* WrbwT = (_Float16*)alloc(1024ull * 1024 * 2);
  _Float16* WoT   = (_Float16*)alloc(1024ull * 1024 * 2);
  float*    heads = (float*)   alloc(2048ull * 3072 * 4);
  float*    rfw   = (float*)   alloc(1024ull * 1024 * 4);
  float*    rbw   = (float*)   alloc(1024ull * 1024 * 4);
  _Float16* q1    = (_Float16*)alloc(32ull * 1024 * 64 * 2);
  _Float16* q2    = (_Float16*)alloc(32ull * 1024 * 64 * 2);
  _Float16* khb   = (_Float16*)alloc(32ull * 1024 * 64 * 2);
  _Float16* vTb   = (_Float16*)alloc(32ull * 64 * 1024 * 2);
  _Float16* rkb   = (_Float16*)alloc(16ull * 2048 * 64 * 2);
  // heads (25 MB) is dead after prep_qkv: alias vec_f16 (4 MB) + attn_out (8 MB) into it
  _Float16* vecf16 = (_Float16*)heads;
  float*    attn   = heads + 2048ull * 1024;   // 8 MB past start of heads

  // 1) precision staging
  cvt_f16_kernel  <<<8192, 256, 0, stream>>>(w, wf16, 2048 * 1024);
  cvt_f16_kernel  <<<4096, 256, 0, stream>>>(r, rf16, 1024 * 1024);
  cvt_f16_T_kernel<<<12288, 256, 0, stream>>>(Wqkv, WqkvT, 1024, 3072);
  cvt_f16_T_kernel<<<4096, 256, 0, stream>>>(Wrfw, WrfwT, 1024, 1024);
  cvt_f16_T_kernel<<<4096, 256, 0, stream>>>(Wrbw, WrbwT, 1024, 1024);
  cvt_f16_T_kernel<<<4096, 256, 0, stream>>>(Wo,   WoT,   1024, 1024);

  // 2) projections (WMMA GEMMs, 16x64 C tile per wave)
  gemm_f16_wmma<<<dim3(6, 128), 256, 0, stream>>>(wf16, WqkvT, heads, 2048, 3072, 1024);
  gemm_f16_wmma<<<dim3(2,  64), 256, 0, stream>>>(rf16, WrfwT, rfw,   1024, 1024, 1024);
  gemm_f16_wmma<<<dim3(2,  64), 256, 0, stream>>>(rf16, WrbwT, rbw,   1024, 1024, 1024);

  // 3) head split + r_k gather
  prep_qkv_kernel<<<8192, 256, 0, stream>>>(heads, rwb, rrb, q1, q2, khb, vTb);
  prep_rk_kernel <<<8192, 256, 0, stream>>>(rfw, rbw, rkb);

  // 4) fused rel-shift flash attention  (2048 wave-tasks / 8 waves per block)
  attn_flash_kernel<<<256, 256, 0, stream>>>(q1, q2, khb, vTb, rkb, mask, vecf16);

  // 5) output projection + residual layernorm
  gemm_f16_wmma<<<dim3(2, 128), 256, 0, stream>>>(vecf16, WoT, attn, 2048, 1024, 1024);
  add_ln_kernel<<<2048, 256, 0, stream>>>(w, attn, lng, lnb, out);
}
